// DeepSeekSparseAttention_53386443489534
// MI455X (gfx1250) — matrix operand order repaired and checked by reading, not compiled, not run on hardware
//
#include <hip/hip_runtime.h>
#include <hip/hip_bf16.h>

// ---------------------------------------------------------------------------
// DeepSeek-style sparse attention for MI455X (gfx1250, wave32, WMMA + TDM).
// Pipeline:
//   f32->f16 convert -> TDM-staged WMMA GEMMs (Q,K,V) -> scores -> top-512
//   -> V^T / gathered K_t,V_t^T -> 3-branch WMMA flash attention -> WMMA out GEMM
// ---------------------------------------------------------------------------

typedef __attribute__((ext_vector_type(16))) _Float16 v16h;
typedef __attribute__((ext_vector_type(8)))  float    v8f;
typedef unsigned int v4u __attribute__((ext_vector_type(4)));
typedef int          v8i __attribute__((ext_vector_type(8)));
typedef int          v4i __attribute__((ext_vector_type(4)));

union FragH { v16h v; uint4 q[2]; };

#define DMODEL 1024
#define NHEADS 16
#define HEADD  64
#define SEQN   2048
#define BATCH  2
#define NTOK   (BATCH * SEQN)     // 4096
#define WIN    128
#define KTOP   512
#define NGLB   64

// ------------------------------ f32 -> f16 --------------------------------
__global__ void k_cvt(const float* __restrict__ src, _Float16* __restrict__ dst, int n4) {
  int i = blockIdx.x * 256 + threadIdx.x;
  if (i < n4) {
    float4 f = reinterpret_cast<const float4*>(src)[i];
    union { _Float16 h[4]; short4 s; } u;
    u.h[0] = (_Float16)f.x; u.h[1] = (_Float16)f.y;
    u.h[2] = (_Float16)f.z; u.h[3] = (_Float16)f.w;
    reinterpret_cast<short4*>(dst)[i] = u.s;
  }
}

// --------------------- TDM: 128x32 f16 tile -> LDS -------------------------
// Loads a 128-row x 32-element f16 tile (row stride `stride` elements) into
// LDS at byte offset lds_off, padding each 64B row to 80B (pad_interval=16 DW,
// pad_amount=4 DW) to match the padded LDS layout used by the WMMA fragments.
__device__ __forceinline__ void tdm_tile_load(const _Float16* gsrc, unsigned lds_off,
                                              int stride) {
  unsigned long long ga = (unsigned long long)(size_t)(const void*)gsrc;
  v4u g0;
  g0[0] = 1u;                                             // count=1 (valid D#)
  g0[1] = lds_off;                                        // lds_addr (bytes)
  g0[2] = (unsigned)ga;                                   // global_addr[31:0]
  g0[3] = (unsigned)((ga >> 32) & 0x01FFFFFFu) | (2u << 30); // addr[56:32] | type=2
  v8i g1;
  g1[0] = (int)((1u << 16)        // data_size = 2 bytes
              | (1u << 20)        // pad_enable
              | (3u << 22)        // pad_interval: 16 DWORDs (one 64B row)
              | (3u << 25));      // pad_amount:   4 DWORDs (16B -> 80B stride)
  g1[1] = (int)(32u << 16);       // tensor_dim0 = 32 (lo16)
  g1[2] = (int)(128u << 16);      // tensor_dim0 hi=0 | tensor_dim1 = 128 (lo16)
  g1[3] = (int)(32u << 16);       // tensor_dim1 hi=0 | tile_dim0 = 32
  g1[4] = 128;                    // tile_dim1 = 128 | tile_dim2 = 0
  g1[5] = stride;                 // tensor_dim0_stride[31:0] (elements)
  g1[6] = 0;                      // stride hi | tensor_dim1_stride lo
  g1[7] = 0;
  v4i gz; gz[0] = gz[1] = gz[2] = gz[3] = 0;              // groups 2/3: unused dims
#if defined(__clang_major__) && (__clang_major__ >= 23)
  v8i gz8;
  gz8[0]=gz8[1]=gz8[2]=gz8[3]=gz8[4]=gz8[5]=gz8[6]=gz8[7]=0;
  __builtin_amdgcn_tensor_load_to_lds(g0, g1, gz, gz, gz8, 0);
#else
  __builtin_amdgcn_tensor_load_to_lds(g0, g1, gz, gz, 0);
#endif
}

// ------------------------- WMMA GEMM: C = A * B^T + bias -------------------
// A: [M,K] f16 row-major. Bw: [N,K] f16 row-major (x@W.T layout).
// Tile 128x128x32, 256 threads (8 waves), each wave: 4x2 16x16 acc tiles.
// Tiles staged to LDS by the Tensor Data Mover, double-buffered: one TDM
// issue + one s_wait_tensorcnt + one barrier per K-step.
template <typename OutT>
__global__ __launch_bounds__(256) void k_gemm(const _Float16* __restrict__ A,
                                              const _Float16* __restrict__ Bw,
                                              const float* __restrict__ bias,
                                              OutT* __restrict__ C,
                                              int M, int Nn, int K) {
  __shared__ _Float16 As[2][128 * 40];   // 80B padded stride (16B-aligned rows)
  __shared__ _Float16 Bs[2][128 * 40];
  const int tid  = threadIdx.x;
  const int wave = tid >> 5;
  const int lane = tid & 31;
  const int lh   = lane & 15;
  const int lg   = lane >> 4;               // 0/1 half-wave group
  const int m0   = blockIdx.x * 128;
  const int n0   = blockIdx.y * 128;
  const int wr   = wave & 1;                // 2 row sub-tiles of 64
  const int wc   = wave >> 1;               // 4 col sub-tiles of 32

  v8f acc[4][2];
#pragma unroll
  for (int i = 0; i < 4; ++i)
#pragma unroll
    for (int j = 0; j < 2; ++j)
#pragma unroll
      for (int r = 0; r < 8; ++r) acc[i][j][r] = 0.0f;

  const _Float16* Ab = A  + (size_t)m0 * K;
  const _Float16* Bb = Bw + (size_t)n0 * K;

  if (tid == 0) {
    tdm_tile_load(Ab, (unsigned)(size_t)(void*)&As[0][0], K);
    tdm_tile_load(Bb, (unsigned)(size_t)(void*)&Bs[0][0], K);
    __builtin_amdgcn_s_wait_tensorcnt(0);
  }
  __syncthreads();

  int cb = 0;
  for (int kk = 0; kk < K; kk += 32) {
    const int nb = cb ^ 1;
    if (tid == 0 && (kk + 32) < K) {        // prefetch next tiles via TDM
      tdm_tile_load(Ab + kk + 32, (unsigned)(size_t)(void*)&As[nb][0], K);
      tdm_tile_load(Bb + kk + 32, (unsigned)(size_t)(void*)&Bs[nb][0], K);
    }

    FragH bf[2];
#pragma unroll
    for (int nf = 0; nf < 2; ++nf) {        // B layout: lanes0-15 K0-15, lanes16-31 K16-31
      int brow = wc * 32 + nf * 16 + lh;
      int off  = lg * 16;
      bf[nf].q[0] = *(const uint4*)&Bs[cb][brow * 40 + off];
      bf[nf].q[1] = *(const uint4*)&Bs[cb][brow * 40 + off + 8];
    }
#pragma unroll
    for (int mf = 0; mf < 4; ++mf) {        // A layout: lanes0-15 K0-15, lanes16-31 K16-31
      FragH af;
      int arow = wr * 64 + mf * 16 + lh;
      int off  = lg * 16;
      af.q[0] = *(const uint4*)&As[cb][arow * 40 + off];
      af.q[1] = *(const uint4*)&As[cb][arow * 40 + off + 8];
#pragma unroll
      for (int nf = 0; nf < 2; ++nf)
        acc[mf][nf] = __builtin_amdgcn_wmma_f32_16x16x32_f16(
            false, af.v, false, bf[nf].v, (short)0, acc[mf][nf], false, false);
    }

    if (tid == 0) __builtin_amdgcn_s_wait_tensorcnt(0);
    __syncthreads();
    cb = nb;
  }

#pragma unroll
  for (int mf = 0; mf < 4; ++mf)
#pragma unroll
    for (int nf = 0; nf < 2; ++nf) {
      int col  = n0 + wc * 32 + nf * 16 + lh;
      float bv = bias[col];
#pragma unroll
      for (int r = 0; r < 8; ++r) {
        int row = m0 + wr * 64 + mf * 16 + r + 8 * lg;
        C[(size_t)row * Nn + col] = (OutT)(acc[mf][nf][r] + bv);
      }
    }
}

// ------------------------------- scores -----------------------------------
__global__ __launch_bounds__(256) void k_scores(const float* __restrict__ x,
                                                const float* __restrict__ Ws,
                                                const float* __restrict__ bs,
                                                float* __restrict__ scores) {
  __shared__ float xs[DMODEL];
  int t = blockIdx.x;                        // token 0..4095
  for (int i = threadIdx.x; i < DMODEL; i += 256) xs[i] = x[(size_t)t * DMODEL + i];
  __syncthreads();
  int h  = threadIdx.x >> 4;
  int sl = threadIdx.x & 15;
  float s = 0.f;
  for (int i = sl; i < DMODEL; i += 16) s += xs[i] * Ws[h * DMODEL + i];
  s += __shfl_xor(s, 1); s += __shfl_xor(s, 2);
  s += __shfl_xor(s, 4); s += __shfl_xor(s, 8);
  if (sl == 0) {
    int b = t >> 11, n = t & (SEQN - 1);
    scores[((size_t)(b * NHEADS + h)) * SEQN + n] = s + bs[h];
  }
}

// -------------------- deterministic top-512 per (b,h) ----------------------
__global__ void k_topk(const float* __restrict__ scores, int* __restrict__ topidx) {
  __shared__ unsigned u[SEQN];
  int bh   = blockIdx.x;
  int lane = threadIdx.x;
  const float* s = scores + (size_t)bh * SEQN;
  for (int i = lane; i < SEQN; i += 32) {
    unsigned b = __float_as_uint(s[i]);
    u[i] = (b & 0x80000000u) ? ~b : (b | 0x80000000u);
  }
  __syncthreads();
  unsigned lo = 0u, hi = 0xFFFFFFFFu;
  while (lo < hi) {
    unsigned mid = (unsigned)((((unsigned long long)lo + hi) + 1ull) >> 1);
    int c = 0;
    for (int i = lane; i < SEQN; i += 32) c += (u[i] >= mid) ? 1 : 0;
    c += __shfl_xor(c, 16); c += __shfl_xor(c, 8);
    c += __shfl_xor(c, 4);  c += __shfl_xor(c, 2); c += __shfl_xor(c, 1);
    if (c >= KTOP) lo = mid; else hi = mid - 1u;
  }
  unsigned T = lo;
  int* out = topidx + bh * KTOP;
  unsigned premask = (lane == 31) ? 0x7FFFFFFFu : ((1u << lane) - 1u);
  int base = 0;
  for (int i0 = 0; i0 < SEQN; i0 += 32) {       // strictly greater
    int i = i0 + lane;
    bool p = u[i] > T;
    unsigned bal = __builtin_amdgcn_ballot_w32(p);
    if (p) out[base + __popc(bal & premask)] = i;
    base += __popc(bal);
  }
  for (int i0 = 0; i0 < SEQN && base < KTOP; i0 += 32) {  // ties, in index order
    int i = i0 + lane;
    bool p = (u[i] == T);
    unsigned bal = __builtin_amdgcn_ballot_w32(p);
    int pre = __popc(bal & premask);
    if (p && (base + pre) < KTOP) out[base + pre] = i;
    base += __popc(bal);
    if (base > KTOP) base = KTOP;
  }
}

// ------------------------------ V transpose --------------------------------
__global__ void k_vt(const _Float16* __restrict__ V16, _Float16* __restrict__ VT) {
  size_t i = (size_t)blockIdx.x * 256 + threadIdx.x;
  int n = (int)(i & (SEQN - 1));
  size_t r = i >> 11;
  int d = (int)(r & 63); r >>= 6;
  int h = (int)(r & 15);
  int b = (int)(r >> 4);
  VT[i] = V16[((size_t)(b * SEQN + n)) * DMODEL + h * HEADD + d];
}

// ------------------------ gather top-k K rows / V^T ------------------------
__global__ void k_gather(const _Float16* __restrict__ K16, const _Float16* __restrict__ V16,
                         const int* __restrict__ topidx,
                         _Float16* __restrict__ Kt, _Float16* __restrict__ VTt) {
  size_t i = (size_t)blockIdx.x * 256 + threadIdx.x;   // over 2*16*512*64
  int d = (int)(i & 63);
  size_t r = i >> 6;
  int j = (int)(r & (KTOP - 1)); r >>= 9;
  int h = (int)(r & 15);
  int b = (int)(r >> 4);
  int n = topidx[(b * NHEADS + h) * KTOP + j];
  size_t src = ((size_t)(b * SEQN + n)) * DMODEL + h * HEADD + d;
  Kt [(((size_t)(b * NHEADS + h)) * KTOP + j) * HEADD + d] = K16[src];
  VTt[(((size_t)(b * NHEADS + h)) * HEADD + d) * KTOP + j] = V16[src];
}

// ----------------------- flash-attention branch (WMMA) ---------------------
__device__ __forceinline__ float redmax16(float v) {
  v = fmaxf(v, __shfl_xor(v, 1)); v = fmaxf(v, __shfl_xor(v, 2));
  v = fmaxf(v, __shfl_xor(v, 4)); v = fmaxf(v, __shfl_xor(v, 8));
  return v;
}
__device__ __forceinline__ float redsum16(float v) {
  v += __shfl_xor(v, 1); v += __shfl_xor(v, 2);
  v += __shfl_xor(v, 4); v += __shfl_xor(v, 8);
  return v;
}

// Online-softmax attention of one 16-query tile over keys [keylo, keylo+nkeys).
__device__ __forceinline__ void attn_branch(const FragH* aQ,
                                            const _Float16* __restrict__ Kbase, int kstride,
                                            const _Float16* __restrict__ VTbase, int vtn,
                                            int keylo, int nkeys,
                                            int q0, int band, float sl2e,
                                            _Float16* ldsP, v8f* Osum) {
  const int lane = threadIdx.x & 31;
  const int lh = lane & 15;
  const int lg = lane >> 4;
  v8f Ob[4];
  float rm[8], rl[8];
#pragma unroll
  for (int nt = 0; nt < 4; ++nt)
#pragma unroll
    for (int r = 0; r < 8; ++r) Ob[nt][r] = 0.0f;
#pragma unroll
  for (int r = 0; r < 8; ++r) { rm[r] = -1.0e30f; rl[r] = 0.0f; }

  for (int kb = 0; kb < nkeys; kb += 32) {
    if (kb + 32 < nkeys) {                  // deepen the load pipeline
      __builtin_prefetch((const void*)(Kbase + (size_t)(keylo + kb + 32 + lh) * kstride), 0, 0);
      __builtin_prefetch((const void*)(VTbase + (size_t)lh * vtn + keylo + kb + 32), 0, 0);
    }
    v8f S[2];
    int nc = (nkeys - kb >= 32) ? 2 : 1;
#pragma unroll
    for (int sub = 0; sub < 2; ++sub) {
      if (sub < nc) {
        int kc = keylo + kb + sub * 16;
        v8f s;
#pragma unroll
        for (int r = 0; r < 8; ++r) s[r] = 0.0f;
        const _Float16* krow = Kbase + (size_t)(kc + lh) * kstride;
#pragma unroll
        for (int f = 0; f < 2; ++f) {       // two k-steps over d (0..31, 32..63)
          FragH bk;
          int off = f * 32 + lg * 16;
          bk.q[0] = *(const uint4*)(krow + off);
          bk.q[1] = *(const uint4*)(krow + off + 8);
          s = __builtin_amdgcn_wmma_f32_16x16x32_f16(
              false, aQ[f].v, false, bk.v, (short)0, s, false, false);
        }
        int key = kc + lh;
#pragma unroll
        for (int r = 0; r < 8; ++r) {
          float val = s[r] * sl2e;          // log2-domain scaling
          if (band) {
            int qrow = q0 + r + 8 * lg;
            int dd = key - qrow;
            if (dd < -WIN || dd > WIN) val = -1.0e30f;
          }
          s[r] = val;
        }
        S[sub] = s;
      } else {
#pragma unroll
        for (int r = 0; r < 8; ++r) S[sub][r] = -1.0e30f;
      }
    }
    // online softmax update
#pragma unroll
    for (int r = 0; r < 8; ++r) {
      float mnew = fmaxf(rm[r], redmax16(fmaxf(S[0][r], S[1][r])));
      float alpha = exp2f(rm[r] - mnew);
      rm[r] = mnew;
      rl[r] *= alpha;
#pragma unroll
      for (int nt = 0; nt < 4; ++nt) Ob[nt][r] *= alpha;
      float p0 = (S[0][r] <= -1.0e29f) ? 0.0f : exp2f(S[0][r] - mnew);
      float p1 = (S[1][r] <= -1.0e29f) ? 0.0f : exp2f(S[1][r] - mnew);
      S[0][r] = p0; S[1][r] = p1;
      rl[r] += redsum16(p0 + p1);
    }
    // P (C-layout f32) -> LDS f16 [16 rows][32 keys] -> A fragment
#pragma unroll
    for (int sub = 0; sub < 2; ++sub)
#pragma unroll
      for (int r = 0; r < 8; ++r)
        ldsP[(r + 8 * lg) * 32 + sub * 16 + lh] = (_Float16)S[sub][r];
    FragH aP;
    {
      int off = lg * 16;
      aP.q[0] = *(const uint4*)&ldsP[lh * 32 + off];
      aP.q[1] = *(const uint4*)&ldsP[lh * 32 + off + 8];
    }
    // PV: O[16x64] += P[16x32] @ V^T-frag[32x16] x 4 d-tiles
#pragma unroll
    for (int nt = 0; nt < 4; ++nt) {
      const _Float16* vrow = VTbase + (size_t)(nt * 16 + lh) * vtn + keylo + kb + lg * 16;
      FragH bv;
      bv.q[0] = *(const uint4*)(vrow);
      bv.q[1] = *(const uint4*)(vrow + 8);
      Ob[nt] = __builtin_amdgcn_wmma_f32_16x16x32_f16(
          false, aP.v, false, bv.v, (short)0, Ob[nt], false, false);
    }
  }
#pragma unroll
  for (int r = 0; r < 8; ++r) {
    float inv = (rl[r] > 0.0f) ? (1.0f / rl[r]) : 0.0f;
#pragma unroll
    for (int nt = 0; nt < 4; ++nt) Osum[nt][r] += Ob[nt][r] * inv;
  }
}

// --------------------------- attention kernel ------------------------------
__global__ __launch_bounds__(256) void k_attn(const _Float16* __restrict__ Q16,
                                              const _Float16* __restrict__ K16,
                                              const _Float16* __restrict__ VT,
                                              const _Float16* __restrict__ Kt,
                                              const _Float16* __restrict__ VTt,
                                              _Float16* __restrict__ AO) {
  __shared__ _Float16 ldsP[8][16 * 32];
  const int wave  = threadIdx.x >> 5;
  const int gwave = blockIdx.x * 8 + wave;        // 0..4095
  const int qt = gwave & 127;
  const int h  = (gwave >> 7) & 15;
  const int b  = gwave >> 11;
  const int q0 = qt * 16;
  const int lane = threadIdx.x & 31;
  const int lh = lane & 15;
  const int lg = lane >> 4;
  const float sl2e = 0.125f * 1.44269504088896340736f;   // 1/sqrt(64) * log2(e)

  FragH aQ[2];
  {
    const _Float16* qrow = Q16 + ((size_t)(b * SEQN + q0 + lh)) * DMODEL + h * HEADD;
#pragma unroll
    for (int f = 0; f < 2; ++f) {
      int off = f * 32 + lg * 16;
      aQ[f].q[0] = *(const uint4*)(qrow + off);
      aQ[f].q[1] = *(const uint4*)(qrow + off + 8);
    }
  }
  v8f Osum[4];
#pragma unroll
  for (int nt = 0; nt < 4; ++nt)
#pragma unroll
    for (int r = 0; r < 8; ++r) Osum[nt][r] = 0.0f;

  const _Float16* Kb  = K16 + ((size_t)b * SEQN) * DMODEL + h * HEADD;
  const _Float16* VTb = VT  + ((size_t)(b * NHEADS + h)) * HEADD * SEQN;
  const _Float16* Ktb  = Kt  + ((size_t)(b * NHEADS + h)) * KTOP * HEADD;
  const _Float16* VTtb = VTt + ((size_t)(b * NHEADS + h)) * HEADD * KTOP;

  int lo = q0 - WIN; if (lo < 0) lo = 0;
  int hi = q0 + 16 + WIN; if (hi > SEQN) hi = SEQN;
  attn_branch(aQ, Kb, DMODEL, VTb, SEQN, lo, hi - lo, q0, 1, sl2e, ldsP[wave], Osum); // local band
  attn_branch(aQ, Ktb, HEADD, VTtb, KTOP, 0, KTOP, q0, 0, sl2e, ldsP[wave], Osum);    // top-k
  attn_branch(aQ, Kb, DMODEL, VTb, SEQN, 0, NGLB, q0, 0, sl2e, ldsP[wave], Osum);     // global

#pragma unroll
  for (int nt = 0; nt < 4; ++nt)
#pragma unroll
    for (int r = 0; r < 8; ++r) {
      int row = q0 + r + 8 * lg;
      int col = h * HEADD + nt * 16 + lh;
      AO[((size_t)(b * SEQN + row)) * DMODEL + col] = (_Float16)(Osum[nt][r] * (1.0f / 3.0f));
    }
}

// ------------------------------- launcher ----------------------------------
extern "C" void kernel_launch(void* const* d_in, const int* in_sizes, int n_in,
                              void* d_out, int out_size, void* d_ws, size_t ws_size,
                              hipStream_t stream) {
  const float* x  = (const float*)d_in[0];
  const float* Wq = (const float*)d_in[1];
  const float* bq = (const float*)d_in[2];
  const float* Wk = (const float*)d_in[3];
  const float* bk = (const float*)d_in[4];
  const float* Wv = (const float*)d_in[5];
  const float* bv = (const float*)d_in[6];
  const float* Wo = (const float*)d_in[7];
  const float* bo = (const float*)d_in[8];
  const float* Ws = (const float*)d_in[9];
  const float* bs = (const float*)d_in[10];

  char* ws = (char*)d_ws;
  size_t off = 0;
  auto nxt = [&](size_t bytes) {
    size_t r = off;
    off = (off + bytes + 255) & ~(size_t)255;
    return r;
  };
  _Float16* x16  = (_Float16*)(ws + nxt((size_t)NTOK * DMODEL * 2));
  _Float16* Wq16 = (_Float16*)(ws + nxt((size_t)DMODEL * DMODEL * 2));
  _Float16* Wk16 = (_Float16*)(ws + nxt((size_t)DMODEL * DMODEL * 2));
  _Float16* Wv16 = (_Float16*)(ws + nxt((size_t)DMODEL * DMODEL * 2));
  _Float16* Wo16 = (_Float16*)(ws + nxt((size_t)DMODEL * DMODEL * 2));
  _Float16* Q16  = (_Float16*)(ws + nxt((size_t)NTOK * DMODEL * 2));
  _Float16* K16  = (_Float16*)(ws + nxt((size_t)NTOK * DMODEL * 2));
  _Float16* V16  = (_Float16*)(ws + nxt((size_t)NTOK * DMODEL * 2));
  _Float16* VT   = (_Float16*)(ws + nxt((size_t)NTOK * DMODEL * 2));
  _Float16* Kt   = (_Float16*)(ws + nxt((size_t)BATCH * NHEADS * KTOP * HEADD * 2));
  _Float16* VTt  = (_Float16*)(ws + nxt((size_t)BATCH * NHEADS * KTOP * HEADD * 2));
  float*    scr  = (float*)   (ws + nxt((size_t)BATCH * NHEADS * SEQN * 4));
  int*      tidx = (int*)     (ws + nxt((size_t)BATCH * NHEADS * KTOP * 4));
  _Float16* AO16 = (_Float16*)(ws + nxt((size_t)NTOK * DMODEL * 2));
  (void)ws_size; (void)in_sizes; (void)n_in; (void)out_size;

  // f32 -> f16
  k_cvt<<<(NTOK * DMODEL / 4 + 255) / 256, 256, 0, stream>>>(x, x16, NTOK * DMODEL / 4);
  k_cvt<<<(DMODEL * DMODEL / 4 + 255) / 256, 256, 0, stream>>>(Wq, Wq16, DMODEL * DMODEL / 4);
  k_cvt<<<(DMODEL * DMODEL / 4 + 255) / 256, 256, 0, stream>>>(Wk, Wk16, DMODEL * DMODEL / 4);
  k_cvt<<<(DMODEL * DMODEL / 4 + 255) / 256, 256, 0, stream>>>(Wv, Wv16, DMODEL * DMODEL / 4);
  k_cvt<<<(DMODEL * DMODEL / 4 + 255) / 256, 256, 0, stream>>>(Wo, Wo16, DMODEL * DMODEL / 4);

  // QKV projections (WMMA, TDM-staged tiles)
  dim3 gg(NTOK / 128, DMODEL / 128);
  k_gemm<_Float16><<<gg, 256, 0, stream>>>(x16, Wq16, bq, Q16, NTOK, DMODEL, DMODEL);
  k_gemm<_Float16><<<gg, 256, 0, stream>>>(x16, Wk16, bk, K16, NTOK, DMODEL, DMODEL);
  k_gemm<_Float16><<<gg, 256, 0, stream>>>(x16, Wv16, bv, V16, NTOK, DMODEL, DMODEL);

  // token scores + deterministic top-512 per (b,h)
  k_scores<<<NTOK, 256, 0, stream>>>(x, Ws, bs, scr);
  k_topk<<<BATCH * NHEADS, 32, 0, stream>>>(scr, tidx);

  // V^T and gathered top-k K / V^T
  k_vt<<<(NTOK * DMODEL + 255) / 256, 256, 0, stream>>>(V16, VT);
  k_gather<<<(BATCH * NHEADS * KTOP * HEADD + 255) / 256, 256, 0, stream>>>(
      K16, V16, tidx, Kt, VTt);

  // 3-branch WMMA flash attention
  k_attn<<<(BATCH * NHEADS * (SEQN / 16)) / 8, 256, 0, stream>>>(Q16, K16, VT, Kt, VTt, AO16);

  // output projection (WMMA, f32 out)
  k_gemm<float><<<gg, 256, 0, stream>>>(AO16, Wo16, bo, (float*)d_out, NTOK, DMODEL, DMODEL);
}
